// Kernel_17611448876263927411_77051713290584
// MI455X (gfx1250) — matrix operand order repaired and checked by reading, not compiled, not run on hardware
//
#include <hip/hip_runtime.h>
#include <hip/hip_bf16.h>

// ---------------------------------------------------------------------------
// MI455X (gfx1250, wave32) implementation.
//   t1  = X(8192x3136) @ P1(3136x256)            -> fused with t4 = p4 * t1
//   t3  = grouped 3x3 conv (groups=2)            -> implicit GEMM, bf16 out
//                                                   stored transposed t3t[n][s][c]
//   t5  = t4 @ p5                                -> wave reduction
//   t7  = (t5 . x) / sqrt(C)                     -> GEMV
//   out = (t4 @ t3)/sqrt(C) + broadcast(t7)      -> GEMM, async-to-LDS copies
// All matrix stages: v_wmma_f32_16x16x32_bf16, f32 accumulation, double-
// buffered LDS (A row-major padded, B transposed/padded so all fragment
// reads are ds_load_b128).  k_t6out uses global_load_async_to_lds_b128
// (ASYNCcnt) when the toolchain exposes it.
// ---------------------------------------------------------------------------

typedef __attribute__((ext_vector_type(16))) __bf16 v16bf;
typedef __attribute__((ext_vector_type(8)))  float  v8f;

#define TM 128   // block tile M
#define TN 64    // block tile N
#define TK 32    // K chunk (matches 16x16x32 WMMA)
#define LPA 40   // padded LDS row stride for A (elements; 80B, 16B-aligned)
#define LPB 40   // padded LDS row stride for transposed B

union Pk8 { __bf16 h[8]; uint4 u; };

#if defined(__AMDGCN__) && __has_builtin(__builtin_amdgcn_global_load_async_to_lds_b128)
#define HAS_ASYNC_LDS 1
// Toolchain signature (from clang diagnostic): param0 is a global-AS pointer
// to a 16B int vector, param1 the LDS-AS counterpart, then two imm ints.
typedef int v4i_t __attribute__((vector_size(16)));
typedef __attribute__((address_space(1))) v4i_t gv4i;
typedef __attribute__((address_space(3))) v4i_t lv4i;
static __device__ __forceinline__ void async_cp_b128(const void* g, void* l) {
  __builtin_amdgcn_global_load_async_to_lds_b128((gv4i*)g, (lv4i*)l, 0, 0);
}
static __device__ __forceinline__ void async_wait0() {
#if __has_builtin(__builtin_amdgcn_s_wait_asynccnt)
  __builtin_amdgcn_s_wait_asynccnt(0);
#else
  asm volatile("s_wait_asynccnt 0" ::: "memory");
#endif
}
#else
#define HAS_ASYNC_LDS 0
#endif

static __device__ __forceinline__ v8f wmma_bf16(v16bf a, v16bf b, v8f c) {
  // 8 args: (neg_a, A, neg_b, B, c_mod, C, reuse_a, reuse_b)
  return __builtin_amdgcn_wmma_f32_16x16x32_bf16(false, a, false, b,
                                                 (short)0, c, false, false);
}

// A fragment: 16x32 bf16 tile, row-major in LDS (row stride LPA).
// k order used here (same for B): lane = half*16 + r ; element j -> K = j + 16*half
// => one contiguous 16-element run starting at k = 16*half (two b128 loads).
static __device__ __forceinline__ v16bf load_frag_a(const __bf16* As, int m0, int lane) {
  const int r = lane & 15, half = lane >> 4;
  const __bf16* row = As + (m0 + r) * LPA;
  union { uint4 u[2]; v16bf v; } cvt;
  cvt.u[0] = *(const uint4*)(row + half * 16);
  cvt.u[1] = *(const uint4*)(row + half * 16 + 8);
  return cvt.v;
}

// B fragment from transposed LDS tile Bt[col][k] (row stride LPB).
// k order used here (same as A): lane = half*16 + col ; element j -> K = j + 16*half
// => one contiguous 16-element run starting at k = 16*half.
static __device__ __forceinline__ v16bf load_frag_bt(const __bf16* Bt, int n0, int lane) {
  const int col = lane & 15, half = lane >> 4;
  const uint4* src = (const uint4*)(Bt + (n0 + col) * LPB + half * 16);
  union { uint4 u[2]; v16bf v; } cvt;
  cvt.u[0] = src[0];
  cvt.u[1] = src[1];
  return cvt.v;
}

// One K-chunk of the 128x64 block tile: wave (wm,wn) owns a 32x32 sub-tile.
static __device__ __forceinline__ void tile_mma(const __bf16* As, const __bf16* Bt,
                                                v8f acc[2][2], int wm, int wn, int lane) {
  v16bf a0 = load_frag_a(As, wm * 32 + 0,  lane);
  v16bf a1 = load_frag_a(As, wm * 32 + 16, lane);
  v16bf b0 = load_frag_bt(Bt, wn * 32 + 0,  lane);
  v16bf b1 = load_frag_bt(Bt, wn * 32 + 16, lane);
  acc[0][0] = wmma_bf16(a0, b0, acc[0][0]);
  acc[0][1] = wmma_bf16(a0, b1, acc[0][1]);
  acc[1][0] = wmma_bf16(a1, b0, acc[1][0]);
  acc[1][1] = wmma_bf16(a1, b1, acc[1][1]);
}

// ---------------------------------------------------------------------------
// Kernel 1: t1 = X @ P1, fused t4 = p4 * t1, stored as bf16 (t4bf[n][a][b]).
//   grid = (64 mtiles, 4 ntiles), block = 256 threads (8 waves).
// ---------------------------------------------------------------------------
__global__ __launch_bounds__(256) void k_t1t4(const float* __restrict__ x,
                                              const float* __restrict__ p1,
                                              const float* __restrict__ p4,
                                              __bf16* __restrict__ t4bf) {
  __shared__ __align__(16) __bf16 As[2][TM * LPA];
  __shared__ __align__(16) __bf16 Bt[2][TN * LPB];

  const int tid  = threadIdx.x;
  const int lane = tid & 31;
  const int wv   = tid >> 5;
  const int wm   = wv & 3, wn = wv >> 2;
  const int m_base = blockIdx.x * TM;
  const int n_base = blockIdx.y * TN;
  const int Kdim = 3136;
  const int NK = Kdim / TK;   // 98

  const int ar = tid >> 1;            // A: 2 threads/row, 16 consecutive k
  const int ak = (tid & 1) * 16;
  const int br = tid >> 3;            // B: 8 threads/k-row, 8 consecutive cols
  const int bc = (tid & 7) * 8;

  const float* gaBase = x + (size_t)(m_base + ar) * Kdim + ak;
  const float* gbBase = p1 + (size_t)br * 256 + n_base + bc;

  v8f acc[2][2] = {};
  float4 ra[4];
  float4 rb[2];

  auto g2r = [&](int kt) {
    const float4* ga = (const float4*)(gaBase + kt * TK);
    ra[0] = ga[0]; ra[1] = ga[1]; ra[2] = ga[2]; ra[3] = ga[3];
    const float4* gb = (const float4*)(gbBase + (size_t)kt * TK * 256);
    rb[0] = gb[0]; rb[1] = gb[1];
  };
  auto r2s = [&](int buf) {
    Pk8 q0, q1;
#pragma unroll
    for (int v = 0; v < 2; ++v) {
      q0.h[v * 4 + 0] = (__bf16)ra[v].x; q0.h[v * 4 + 1] = (__bf16)ra[v].y;
      q0.h[v * 4 + 2] = (__bf16)ra[v].z; q0.h[v * 4 + 3] = (__bf16)ra[v].w;
      q1.h[v * 4 + 0] = (__bf16)ra[v + 2].x; q1.h[v * 4 + 1] = (__bf16)ra[v + 2].y;
      q1.h[v * 4 + 2] = (__bf16)ra[v + 2].z; q1.h[v * 4 + 3] = (__bf16)ra[v + 2].w;
    }
    *(uint4*)&As[buf][ar * LPA + ak]     = q0.u;
    *(uint4*)&As[buf][ar * LPA + ak + 8] = q1.u;
    const float* fb = (const float*)rb;
#pragma unroll
    for (int j = 0; j < 8; ++j)
      Bt[buf][(bc + j) * LPB + br] = (__bf16)fb[j];
  };

  g2r(0);
  r2s(0);
  __syncthreads();

  for (int kt = 0; kt < NK; ++kt) {
    const int nxt = kt + 1;
    if (nxt < NK) g2r(nxt);
    if (kt + 4 < NK) {
      __builtin_prefetch(gaBase + (kt + 4) * TK, 0, 1);
      __builtin_prefetch(gbBase + (size_t)(kt + 4) * TK * 256, 0, 1);
    }
    tile_mma(As[kt & 1], Bt[kt & 1], acc, wm, wn, lane);
    if (nxt < NK) r2s(nxt & 1);
    __syncthreads();
  }

  // Epilogue: t4[m, col] = p4[m%256, col] * t1 ; store bf16.
  const int colb = lane & 15, half = lane >> 4;
#pragma unroll
  for (int fi = 0; fi < 2; ++fi) {
#pragma unroll
    for (int fj = 0; fj < 2; ++fj) {
      const int ncol = n_base + wn * 32 + fj * 16 + colb;
#pragma unroll
      for (int i = 0; i < 8; ++i) {
        const int m = m_base + wm * 32 + fi * 16 + half * 8 + i;
        const float scale = p4[(m & 255) * 256 + ncol];
        t4bf[(size_t)m * 256 + ncol] = (__bf16)(acc[fi][fj][i] * scale);
      }
    }
  }
}

// ---------------------------------------------------------------------------
// Kernel 2: grouped 3x3 conv (groups=2) as implicit GEMM.
//   Output stored TRANSPOSED: t3t[n][s][c]  (c contiguous) so that the
//   consumer GEMM can async-copy contiguous 16B runs and the epilogue here
//   emits packed b128 stores (8 consecutive channels per lane).
//   grid = (49 s-tiles, 2 groups, 32 batch), block = 256.
// ---------------------------------------------------------------------------
__global__ __launch_bounds__(256) void k_conv(const float* __restrict__ x,
                                              const float* __restrict__ w,
                                              __bf16* __restrict__ t3t) {
  __shared__ __align__(16) __bf16 As[2][TM * LPA];
  __shared__ __align__(16) __bf16 Bt[2][TN * LPB];

  const int tid  = threadIdx.x;
  const int lane = tid & 31;
  const int wv   = tid >> 5;
  const int wm   = wv & 3, wn = wv >> 2;
  const int st = blockIdx.x, g = blockIdx.y, n = blockIdx.z;
  const int s_base = st * TN;
  const int HWs = 3136;
  const int NK = 36;

  const int ar = tid >> 1;
  const int ak = (tid & 1) * 16;
  const int br = tid >> 3;
  const int bc = (tid & 7) * 8;

  v8f acc[2][2] = {};
  float fa[16];
  float fb[8];

  auto g2r = [&](int kt) {
    const int tap = kt >> 2;          // 0..8 : kh*3+kw
    const int ci0 = (kt & 3) * 32;    // in-channel base within group
    const int kh = tap / 3, kw = tap % 3;
    const float* wbase = w + ((size_t)(g * 128 + ar) * 128 + ci0 + ak) * 9 + tap;
#pragma unroll
    for (int j = 0; j < 16; ++j) fa[j] = wbase[j * 9];
    const int ci = g * 128 + ci0 + br;
    const float* xb = x + ((size_t)n * 256 + ci) * HWs;
#pragma unroll
    for (int j = 0; j < 8; ++j) {
      const int s = s_base + bc + j;
      const int h = s / 56, ww = s % 56;
      const int hh = h + kh - 1, w2 = ww + kw - 1;
      float v = 0.f;
      if (hh >= 0 && hh < 56 && w2 >= 0 && w2 < 56) v = xb[hh * 56 + w2];
      fb[j] = v;
    }
  };
  auto r2s = [&](int buf) {
    Pk8 q0, q1;
#pragma unroll
    for (int j = 0; j < 8; ++j) { q0.h[j] = (__bf16)fa[j]; q1.h[j] = (__bf16)fa[j + 8]; }
    *(uint4*)&As[buf][ar * LPA + ak]     = q0.u;
    *(uint4*)&As[buf][ar * LPA + ak + 8] = q1.u;
#pragma unroll
    for (int j = 0; j < 8; ++j)
      Bt[buf][(bc + j) * LPB + br] = (__bf16)fb[j];
  };

  g2r(0);
  r2s(0);
  __syncthreads();

  for (int kt = 0; kt < NK; ++kt) {
    const int nxt = kt + 1;
    if (nxt < NK) g2r(nxt);
    tile_mma(As[kt & 1], Bt[kt & 1], acc, wm, wn, lane);
    if (nxt < NK) r2s(nxt & 1);
    __syncthreads();
  }

  // Store t3t[n][s][c]: per (fi,fj) each lane owns 8 consecutive channels.
  const int colb = lane & 15, half = lane >> 4;
#pragma unroll
  for (int fi = 0; fi < 2; ++fi) {
#pragma unroll
    for (int fj = 0; fj < 2; ++fj) {
      const int s = s_base + wn * 32 + fj * 16 + colb;
      Pk8 q;
#pragma unroll
      for (int i = 0; i < 8; ++i) q.h[i] = (__bf16)acc[fi][fj][i];
      const int m = g * 128 + wm * 32 + fi * 16 + half * 8;
      *(uint4*)(t3t + ((size_t)n * HWs + s) * 256 + m) = q.u;
    }
  }
}

// ---------------------------------------------------------------------------
// Kernel 3: t5[n,a] = sum_e t4[n,a,e] * p5[e].  One wave per (n,a).
// ---------------------------------------------------------------------------
__global__ __launch_bounds__(256) void k_t5(const __bf16* __restrict__ t4bf,
                                            const float* __restrict__ p5,
                                            float* __restrict__ t5) {
  const int gw   = (blockIdx.x * 256 + threadIdx.x) >> 5;  // 0..8191
  const int lane = threadIdx.x & 31;
  const __bf16* row = t4bf + (size_t)gw * 256;
  float sum = 0.f;
#pragma unroll
  for (int e = 0; e < 256; e += 32) sum += (float)row[e + lane] * p5[e + lane];
#pragma unroll
  for (int off = 16; off > 0; off >>= 1) sum += __shfl_down(sum, off, 32);
  if (lane == 0) t5[gw] = sum;
}

// ---------------------------------------------------------------------------
// Kernel 4: t7[n,s] = (1/16) * sum_c t5[n,c] * x[n,c,s].
//   grid = (13 s-chunks, 32 batch), block = 256; coalesced over s.
// ---------------------------------------------------------------------------
__global__ __launch_bounds__(256) void k_t7(const float* __restrict__ x,
                                            const float* __restrict__ t5,
                                            float* __restrict__ t7) {
  __shared__ float s5[256];
  const int n = blockIdx.y;
  const int s = blockIdx.x * 256 + threadIdx.x;
  s5[threadIdx.x] = t5[n * 256 + threadIdx.x];
  __syncthreads();
  if (s >= 3136) return;
  const float* xb = x + (size_t)n * 256 * 3136 + s;
  float sum = 0.f;
#pragma unroll 8
  for (int c = 0; c < 256; ++c) sum += s5[c] * xb[(size_t)c * 3136];
  t7[n * 3136 + s] = sum * 0.0625f;   // 1/sqrt(256)
}

// ---------------------------------------------------------------------------
// Kernel 5: out[n,a,s] = (1/16) * (t4[n] @ t3[n])[a,s] + t7[n,s].
//   Per n: M=256, K=256 (bf16), N=3136.  grid = (49, 2 mtiles, 32).
//   A = t4bf[n][a][b] (b-contiguous), B = t3t[n][s][c] (c-contiguous):
//   both tiles are pure contiguous copies -> global_load_async_to_lds_b128.
// ---------------------------------------------------------------------------
__global__ __launch_bounds__(256) void k_t6out(const __bf16* __restrict__ t4bf,
                                               const __bf16* __restrict__ t3t,
                                               const float* __restrict__ t7,
                                               float* __restrict__ out) {
  __shared__ __align__(16) __bf16 As[2][TM * LPA];
  __shared__ __align__(16) __bf16 Bt[2][TN * LPB];

  const int tid  = threadIdx.x;
  const int lane = tid & 31;
  const int wv   = tid >> 5;
  const int wm   = wv & 3, wn = wv >> 2;
  const int st = blockIdx.x, mt = blockIdx.y, n = blockIdx.z;
  const int s_base = st * TN;
  const int m_base = mt * TM;
  const int HWs = 3136;
  const int NK = 8;

  // A: 2 threads/row, 16 consecutive k each (two b128s).
  const int ar = tid >> 1;
  const int ak = (tid & 1) * 16;
  // B: 4 threads/column, 8 consecutive k each (one b128) into Bt[col][k].
  const int sc = tid >> 2;
  const int kc = (tid & 3) * 8;

  const __bf16* gaBase = t4bf + (size_t)n * 65536 + (size_t)(m_base + ar) * 256 + ak;
  const __bf16* gbBase = t3t + ((size_t)n * HWs + s_base + sc) * 256 + kc;

  v8f acc[2][2] = {};

#if HAS_ASYNC_LDS
  auto g2s_async = [&](int kt, int buf) {
    async_cp_b128(gaBase + kt * TK,     &As[buf][ar * LPA + ak]);
    async_cp_b128(gaBase + kt * TK + 8, &As[buf][ar * LPA + ak + 8]);
    async_cp_b128(gbBase + kt * TK,     &Bt[buf][sc * LPB + kc]);
  };
  g2s_async(0, 0);
  async_wait0();
  __syncthreads();
  for (int kt = 0; kt < NK; ++kt) {
    const int nxt = kt + 1;
    if (nxt < NK) g2s_async(nxt, nxt & 1);
    tile_mma(As[kt & 1], Bt[kt & 1], acc, wm, wn, lane);
    if (nxt < NK) async_wait0();
    __syncthreads();
  }
#else
  uint4 rau[2];
  uint4 rbu;
  auto g2r = [&](int kt) {
    const uint4* ga = (const uint4*)(gaBase + kt * TK);
    rau[0] = ga[0]; rau[1] = ga[1];
    rbu = *(const uint4*)(gbBase + kt * TK);
  };
  auto r2s = [&](int buf) {
    *(uint4*)&As[buf][ar * LPA + ak]     = rau[0];
    *(uint4*)&As[buf][ar * LPA + ak + 8] = rau[1];
    *(uint4*)&Bt[buf][sc * LPB + kc]     = rbu;
  };
  g2r(0);
  r2s(0);
  __syncthreads();
  for (int kt = 0; kt < NK; ++kt) {
    const int nxt = kt + 1;
    if (nxt < NK) g2r(nxt);
    tile_mma(As[kt & 1], Bt[kt & 1], acc, wm, wn, lane);
    if (nxt < NK) r2s(nxt & 1);
    __syncthreads();
  }
#endif

  const float inv = 0.0625f;  // 1/sqrt(256)
  const int colb = lane & 15, half = lane >> 4;
#pragma unroll
  for (int fi = 0; fi < 2; ++fi) {
#pragma unroll
    for (int fj = 0; fj < 2; ++fj) {
      const int s = s_base + wn * 32 + fj * 16 + colb;
      const float t7v = t7[n * HWs + s];
#pragma unroll
      for (int i = 0; i < 8; ++i) {
        const int m = m_base + wm * 32 + fi * 16 + half * 8 + i;
        out[((size_t)n * 256 + m) * HWs + s] = acc[fi][fj][i] * inv + t7v;
      }
    }
  }
}

// ---------------------------------------------------------------------------
extern "C" void kernel_launch(void* const* d_in, const int* in_sizes, int n_in,
                              void* d_out, int out_size, void* d_ws, size_t ws_size,
                              hipStream_t stream) {
  (void)in_sizes; (void)n_in; (void)out_size; (void)ws_size;
  const float* x  = (const float*)d_in[0];
  const float* p1 = (const float*)d_in[1];
  const float* cw = (const float*)d_in[2];
  const float* p4 = (const float*)d_in[3];
  const float* p5 = (const float*)d_in[4];
  float* out = (float*)d_out;

  char* ws = (char*)d_ws;
  // Workspace layout (bytes):
  //   t4bf : 32*256*256*2   =  4,194,304   t4bf[n][a][b]
  //   t3t  : 32*3136*256*2  = 51,380,224   t3t[n][s][c]
  //   t5   : 32*256*4       =     32,768
  //   t7   : 32*3136*4      =    401,408
  __bf16* t4bf = (__bf16*)ws;
  __bf16* t3t  = (__bf16*)(ws + 4194304u);
  float*  t5   = (float*)(ws + 4194304u + 51380224u);
  float*  t7   = (float*)(ws + 4194304u + 51380224u + 32768u);

  k_t1t4 <<<dim3(64, 4, 1),  256, 0, stream>>>(x, p1, p4, t4bf);
  k_conv <<<dim3(49, 2, 32), 256, 0, stream>>>(x, cw, t3t);
  k_t5   <<<dim3(1024, 1, 1), 256, 0, stream>>>(t4bf, p5, t5);
  k_t7   <<<dim3(13, 32, 1), 256, 0, stream>>>(x, t5, t7);
  k_t6out<<<dim3(49, 2, 32), 256, 0, stream>>>(t4bf, t3t, t7, out);
}
